// ConvLSTMNeck_41652592837358
// MI455X (gfx1250) — compile-verified, matrix operand order checked by reading, not run on hardware
//
#include <hip/hip_runtime.h>

// ---------------- problem constants ----------------
#define LNUM 2
#define BATCH 8
#define TSTEPS 12
#define CIN 96
#define HID 96
#define HW 48
#define SPAT (HW * HW)              // 2304
#define ZC (CIN + HID)              // 192 (NHWC channel dim of concat buffer)
#define GM (4 * HID)                // 384  (M of GEMM, gate-interleaved: m = ch*4+gate)
#define KDIM (ZC * 9)               // 1728 (K of GEMM, tap-major => contiguous)
#define BK 32
#define KSTEPS (KDIM / BK)          // 54 = 9 taps * 6 channel-chunks
#define BM 128
#define BN 128
#define AST 40                      // LDS row stride (bf16): 80B, conflict-free mod 256B
#define BST 40
#define CST 132                     // LDS row stride (f32) for C tile

// output layout: all_hidden [L,B,T,HID,H,W] ++ h_f [L,B,HID,H,W] ++ c_f [...]
#define OFF_HF ((size_t)LNUM * BATCH * TSTEPS * HID * SPAT)   // 42,467,328
#define OFF_CF (OFF_HF + (size_t)LNUM * BATCH * HID * SPAT)   // 46,006,272

typedef __attribute__((ext_vector_type(16))) __bf16 v16bf;
typedef __attribute__((ext_vector_type(8)))  float  v8f;

__device__ __forceinline__ unsigned short f2bf(float f) {
  unsigned int u = __float_as_uint(f);
  u += 0x7fffu + ((u >> 16) & 1u);   // round-to-nearest-even
  return (unsigned short)(u >> 16);
}

__device__ __forceinline__ float sigf(float x) {
  return 1.0f / (1.0f + __expf(-x));
}

// -------- weight f32 OIHW -> bf16, tap-major K, gate-interleaved M --------
// dst[l][m'][tap*ZC+ci] with m' = ch*4 + gate;  src o = gate*HID + ch
__global__ void __launch_bounds__(256)
wconv_kernel(const float* __restrict__ w, unsigned short* __restrict__ wbf) {
  int idx = blockIdx.x * 256 + threadIdx.x;     // over LNUM*GM*KDIM
  if (idx >= LNUM * GM * KDIM) return;
  int l   = idx / (GM * KDIM);
  int rem = idx - l * (GM * KDIM);
  int m   = rem / KDIM;
  int k   = rem - m * KDIM;
  int tap = k / ZC;
  int ci  = k - tap * ZC;
  int ch  = m >> 2;
  int g   = m & 3;
  wbf[idx] = f2bf(w[(((size_t)l * GM + g * HID + ch) * ZC + ci) * 9 + tap]);
}

// ---------------- init c-state + h-half of concat buffers (NHWC) ----------------
__global__ void __launch_bounds__(256)
init_state_kernel(const float* __restrict__ h0, const float* __restrict__ c0,
                  unsigned short* __restrict__ z0, unsigned short* __restrict__ z1,
                  float* __restrict__ c_state) {
  int idx = blockIdx.x * 256 + threadIdx.x;          // over L*B*HID*SPAT
  if (idx >= LNUM * BATCH * HID * SPAT) return;
  int l   = idx / (BATCH * HID * SPAT);
  int rem = idx - l * (BATCH * HID * SPAT);
  int b   = rem / (HID * SPAT);
  int r2  = rem - b * (HID * SPAT);
  int ch  = r2 / SPAT;
  int p   = r2 - ch * SPAT;
  c_state[idx] = c0[idx];
  unsigned short* z = (l == 0) ? z0 : z1;
  z[((size_t)b * SPAT + p) * ZC + CIN + ch] = f2bf(h0[idx]);
}

// ---------------- pack x_t into x-half of z0 (NHWC) ----------------
__global__ void __launch_bounds__(256)
pack_x_kernel(const float* __restrict__ inp, unsigned short* __restrict__ z0, int t) {
  int idx = blockIdx.x * 256 + threadIdx.x;          // over B*CIN*SPAT
  if (idx >= BATCH * CIN * SPAT) return;
  int b  = idx / (CIN * SPAT);
  int r  = idx - b * (CIN * SPAT);
  int ch = r / SPAT;
  int p  = r - ch * SPAT;
  float v = inp[(((size_t)b * TSTEPS + t) * CIN + ch) * SPAT + p];
  z0[((size_t)b * SPAT + p) * ZC + ch] = f2bf(v);
}

// ---------------- WMMA compute for one K-step ----------------
__device__ __forceinline__ void wmma_step(const unsigned short* __restrict__ lA,
                                          const unsigned short* __restrict__ lB,
                                          int m_base, int n_base, int lane,
                                          v8f acc[2][4]) {
  const int lrow  = lane & 15;
  const int khalf = (lane >> 4) * 16;
  v16bf af0 = *(const v16bf*)&lA[((m_base + 0) * 16 + lrow) * AST + khalf];
  v16bf af1 = *(const v16bf*)&lA[((m_base + 1) * 16 + lrow) * AST + khalf];
#pragma unroll
  for (int j = 0; j < 4; ++j) {
    v16bf bf = *(const v16bf*)&lB[((n_base + j) * 16 + lrow) * BST + khalf];
    acc[0][j] = __builtin_amdgcn_wmma_f32_16x16x32_bf16(
        false, af0, false, bf, (short)0, acc[0][j], false, false);
    acc[1][j] = __builtin_amdgcn_wmma_f32_16x16x32_bf16(
        false, af1, false, bf, (short)0, acc[1][j], false, false);
  }
}

// -------- fused implicit-GEMM conv (WMMA bf16) + LSTM cell epilogue --------
// grid = (SPAT/BN=18, GM/BM=3, B=8), block = 256 (8 waves)
// M tile covers 32 channels x 4 gates (gate-interleaved rows).
__global__ void __launch_bounds__(256, 1)
convlstm_step_kernel(const unsigned short* __restrict__ wl,   // [GM][KDIM] bf16
                     const unsigned short* __restrict__ z,    // [B][48][48][ZC] bf16
                     const float* __restrict__ bias,          // b[l], 384 (i,f,o,g)
                     float* __restrict__ c_state,             // layer slice [B][HID][SPAT]
                     float* __restrict__ out,                 // full d_out
                     unsigned short* __restrict__ z_self,     // this layer concat (NHWC)
                     unsigned short* __restrict__ z_next,     // next layer concat or null
                     int l, int t, int write_final) {
  __shared__ __align__(16) union {
    struct {
      unsigned short a[2][BM * AST];   // 20,480 B
      unsigned short bsh[2][BN * BST]; // 20,480 B
    } stage;
    float c[BM * CST];                 // 67,584 B  (aliased after GEMM phase)
  } lds;

  const int tid  = threadIdx.x;
  const int lane = tid & 31;
  const int wid  = tid >> 5;
  const int b    = blockIdx.z;
  const int m0   = blockIdx.y * BM;
  const int n0   = blockIdx.x * BN;

  const int m_base = (wid & 3) * 2;     // 2 m-subtiles per wave
  const int n_base = (wid >> 2) * 4;    // 4 n-subtiles per wave

  // fills: 128 rows x 32 k each, 16 bf16 (2 x uint4) per thread
  const int row  = tid >> 1;            // 0..127
  const int col  = (tid & 1) * 16;      // 0 or 16
  const unsigned short* ap = wl + (size_t)(m0 + row) * KDIM + col;

  // per-thread halo constants (selected per step by uniform kh/kw)
  const int n_g = n0 + row;
  const int y   = n_g / HW;
  const int x   = n_g - y * HW;
  const unsigned short* zb = z + (size_t)b * SPAT * ZC;

  int  rofs0 = max(y - 1, 0) * HW, rofs1 = y * HW, rofs2 = min(y + 1, HW - 1) * HW;
  int  cofs0 = max(x - 1, 0),      cofs1 = x,      cofs2 = min(x + 1, HW - 1);
  bool oky0 = (y - 1) >= 0, oky2 = (y + 1) < HW;
  bool okx0 = (x - 1) >= 0, okx2 = (x + 1) < HW;

  v8f acc[2][4];
#pragma unroll
  for (int i = 0; i < 2; ++i)
#pragma unroll
    for (int j = 0; j < 4; ++j) acc[i][j] = (v8f){};

  uint4 aR0, aR1, bR0, bR1;

#define LOAD_STEP(s)                                                          \
  do {                                                                        \
    aR0 = *(const uint4*)&ap[(s) * BK];                                       \
    aR1 = *(const uint4*)&ap[(s) * BK + 8];                                   \
    int tap = (s) / 6;          /* uniform */                                 \
    int cc  = ((s) - tap * 6) * BK;                                           \
    int kh  = tap / 3, kw = tap - kh * 3;                                     \
    int  rofs = (kh == 0) ? rofs0 : (kh == 1) ? rofs1 : rofs2;                \
    int  cofs = (kw == 0) ? cofs0 : (kw == 1) ? cofs1 : cofs2;                \
    bool ok   = ((kh == 0) ? oky0 : (kh == 1) ? true : oky2) &&               \
                ((kw == 0) ? okx0 : (kw == 1) ? true : okx2);                 \
    const uint4* src = (const uint4*)&zb[(size_t)(rofs + cofs) * ZC + cc + col]; \
    uint4 t0 = src[0], t1 = src[1];                                           \
    bR0.x = ok ? t0.x : 0u; bR0.y = ok ? t0.y : 0u;                           \
    bR0.z = ok ? t0.z : 0u; bR0.w = ok ? t0.w : 0u;                           \
    bR1.x = ok ? t1.x : 0u; bR1.y = ok ? t1.y : 0u;                           \
    bR1.z = ok ? t1.z : 0u; bR1.w = ok ? t1.w : 0u;                           \
  } while (0)

#define STORE_STEP(buf)                                                       \
  do {                                                                        \
    *(uint4*)&lds.stage.a[buf][row * AST + col]       = aR0;                  \
    *(uint4*)&lds.stage.a[buf][row * AST + col + 8]   = aR1;                  \
    *(uint4*)&lds.stage.bsh[buf][row * BST + col]     = bR0;                  \
    *(uint4*)&lds.stage.bsh[buf][row * BST + col + 8] = bR1;                  \
  } while (0)

  LOAD_STEP(0);
  STORE_STEP(0);
  __syncthreads();

  for (int s = 0; s < KSTEPS - 1; ++s) {
    const int cur = s & 1;
    const int nxt = cur ^ 1;
    LOAD_STEP(s + 1);                        // global loads overlap WMMA below
    if (s + 2 < KSTEPS)                      // prefetch weight stream
      __builtin_prefetch(&ap[(s + 2) * BK], 0, 0);
    wmma_step(lds.stage.a[cur], lds.stage.bsh[cur], m_base, n_base, lane, acc);
    STORE_STEP(nxt);
    __syncthreads();
  }
  wmma_step(lds.stage.a[(KSTEPS - 1) & 1], lds.stage.bsh[(KSTEPS - 1) & 1],
            m_base, n_base, lane, acc);

#undef LOAD_STEP
#undef STORE_STEP

  // ---- stage accumulators into LDS C tile (aliases the staging buffers) ----
  __syncthreads();    // everyone done reading stage LDS
  {
    const int lrow  = lane & 15;
    const int rbase = (lane >> 4) * 8;
#pragma unroll
    for (int i = 0; i < 2; ++i)
#pragma unroll
      for (int j = 0; j < 4; ++j)
#pragma unroll
        for (int r = 0; r < 8; ++r)
          lds.c[((m_base + i) * 16 + rbase + r) * CST +
                (n_base + j) * 16 + lrow] = acc[i][j][r];
  }
  __syncthreads();

  // ---- fused LSTM cell: 32 channels x 128 positions in this tile ----
  const int chl  = tid >> 3;            // 0..31 local channel
  const int ploc = (tid & 7) * 16;      // 16 consecutive positions
  const int chg  = (m0 >> 2) + chl;     // global channel 0..95
  const float bi = bias[chg];
  const float bf = bias[HID + chg];
  const float bo = bias[2 * HID + chg];
  const float bg = bias[3 * HID + chg];

  const float* gi_row = &lds.c[(chl * 4 + 0) * CST + ploc];
  const float* gf_row = &lds.c[(chl * 4 + 1) * CST + ploc];
  const float* go_row = &lds.c[(chl * 4 + 2) * CST + ploc];
  const float* gg_row = &lds.c[(chl * 4 + 3) * CST + ploc];

  float* crow = c_state + ((size_t)b * HID + chg) * SPAT + n0 + ploc;
  float* hrow = out +
      ((((size_t)l * BATCH + b) * TSTEPS + t) * HID + chg) * SPAT + n0 + ploc;

#pragma unroll
  for (int q = 0; q < 16; ++q) {
    float xi = gi_row[q] + bi;
    float xf = gf_row[q] + bf;
    float xo = go_row[q] + bo;
    float xg = gg_row[q] + bg;
    float c_old = crow[q];
    float c_new = sigf(xf) * c_old + sigf(xi) * tanhf(xg);
    float h_new = sigf(xo) * tanhf(c_new);
    crow[q] = c_new;
    hrow[q] = h_new;

    int pg = n0 + ploc + q;                                // global spatial pos
    unsigned short hb = f2bf(h_new);
    size_t znhwc = ((size_t)b * SPAT + pg) * ZC;
    z_self[znhwc + CIN + chg] = hb;                        // recurrent, next t
    if (z_next) z_next[znhwc + chg] = hb;                  // feeds layer l+1

    if (write_final) {
      size_t fo = (((size_t)l * BATCH + b) * HID + chg) * SPAT + pg;
      out[OFF_HF + fo] = h_new;
      out[OFF_CF + fo] = c_new;
    }
  }
}

// ---------------- host-side orchestration ----------------
extern "C" void kernel_launch(void* const* d_in, const int* in_sizes, int n_in,
                              void* d_out, int out_size, void* d_ws, size_t ws_size,
                              hipStream_t stream) {
  const float* inp  = (const float*)d_in[0];
  const float* h0   = (const float*)d_in[1];
  const float* c0   = (const float*)d_in[2];
  const float* Wt   = (const float*)d_in[3];
  const float* bias = (const float*)d_in[4];
  float* out = (float*)d_out;
  char*  ws  = (char*)d_ws;

  const size_t WBF_ELEMS = (size_t)LNUM * GM * KDIM;            // 1,327,104
  const size_t Z_ELEMS   = (size_t)BATCH * SPAT * ZC;           // 3,538,944

  unsigned short* wbf = (unsigned short*)ws;
  unsigned short* z0  = (unsigned short*)(ws + WBF_ELEMS * 2);
  unsigned short* z1  = z0 + Z_ELEMS;
  float* c_state      = (float*)((char*)(z1 + Z_ELEMS));        // [L][B][HID][SPAT]
  (void)in_sizes; (void)n_in; (void)out_size; (void)ws_size;

  {
    int n = (int)WBF_ELEMS;
    wconv_kernel<<<(n + 255) / 256, 256, 0, stream>>>(Wt, wbf);
  }
  {
    int n = LNUM * BATCH * HID * SPAT;
    init_state_kernel<<<(n + 255) / 256, 256, 0, stream>>>(h0, c0, z0, z1, c_state);
  }

  const dim3 ggrid(SPAT / BN, GM / BM, BATCH);   // (18, 3, 8)
  const int  pcnt = BATCH * CIN * SPAT;
  const int  pblk = (pcnt + 255) / 256;

  for (int t = 0; t < TSTEPS; ++t) {
    pack_x_kernel<<<pblk, 256, 0, stream>>>(inp, z0, t);

    // layer 0: conv + cell fused
    convlstm_step_kernel<<<ggrid, 256, 0, stream>>>(
        wbf, z0, bias, c_state, out, z0, z1, 0, t, t == TSTEPS - 1);

    // layer 1: conv + cell fused
    convlstm_step_kernel<<<ggrid, 256, 0, stream>>>(
        wbf + (size_t)GM * KDIM, z1, bias + GM,
        c_state + (size_t)BATCH * HID * SPAT, out,
        z1, (unsigned short*)nullptr, 1, t, t == TSTEPS - 1);
  }
}
